// MultiHeadAttention_56143812493621
// MI455X (gfx1250) — matrix operand order repaired and checked by reading, not compiled, not run on hardware
//
#include <hip/hip_runtime.h>
#include <hip/hip_bf16.h>

// ---------------------------------------------------------------------------
// MI455X (gfx1250) multi-head attention: bf16 WMMA (v_wmma_f32_16x16x32_bf16),
// flash-attention streaming softmax, transposed-score trick so the P matrix
// needs no cross-lane transpose between the QK^T and PV matmuls.
// v2: software-pipelined k-loops, hoisted per-lane base pointers, load/compute
// overlap in the attention loop (loads issued before the softmax VALU chain).
// ---------------------------------------------------------------------------

typedef __attribute__((ext_vector_type(16))) __bf16 bf16x16;
typedef __attribute__((ext_vector_type(8)))  __bf16 bf16x8;
typedef __attribute__((ext_vector_type(8)))  float  f32x8;
typedef __attribute__((ext_vector_type(4)))  float  f32x4;

#define D_MODEL 1024
#define N_HEADS 16
#define D_K     64
#define SEQ     2048
#define BATCH   2
#define ROWS    (BATCH * SEQ)      // 4096

__device__ __forceinline__ f32x8 wmma_bf16(bf16x16 a, bf16x16 b, f32x8 c) {
  // D(16x16 f32) = A(16x32 bf16) * B(32x16 bf16) + C
  return __builtin_amdgcn_wmma_f32_16x16x32_bf16(
      /*neg_a=*/false, a, /*neg_b=*/false, b,
      /*c_mod=*/(short)0, c, /*reuse_a=*/false, /*reuse_b=*/false);
}

__device__ __forceinline__ float lane_bcast(float x, int src_lane) {
  int r = __builtin_amdgcn_ds_bpermute(src_lane << 2, __float_as_int(x));
  return __int_as_float(r);
}

// ---- Fragment layouts (ISA 7.12.2, wave32) ---------------------------------
// A (16x32, bf16): lane L -> row m = L%16; elements e: k = (e/8)*16 + (L/16)*8 + e%8
// B (32x16, bf16): lane L -> col n = L%16; elements e: k = (e/8)*16 + (L/16)*8 + e%8
// C/D (16x16 f32): lane L -> col n = L%16; VGPR v -> row m = v + 8*(L/16)

// Per-lane base pointers (computed once); fragments fetched at element offset k0.
__device__ __forceinline__ const __bf16* a_base_bf16(const __bf16* X, int ld,
                                                     int row0, int lane) {
  return X + (size_t)(row0 + (lane & 15)) * ld + (lane >> 4) * 8;
}
__device__ __forceinline__ bf16x16 a_frag_bf16(const __bf16* p, int k0) {
  bf16x8 lo = *(const bf16x8*)(p + k0);         // k = k0+kh .. +7  (16B aligned)
  bf16x8 hi = *(const bf16x8*)(p + k0 + 16);    // k = k0+16+kh .. +7
  return __builtin_shufflevector(lo, hi, 0,1,2,3,4,5,6,7,8,9,10,11,12,13,14,15);
}

__device__ __forceinline__ const __bf16* b_base_bf16(const __bf16* X, int ld,
                                                     int n0, int lane) {
  // B[k][n] taken from X[n][k] (rows indexed by n)
  return X + (size_t)(n0 + (lane & 15)) * ld + (lane >> 4) * 8;
}
__device__ __forceinline__ bf16x16 b_frag_bf16(const __bf16* p, int k0) {
  bf16x8 lo = *(const bf16x8*)(p + k0);
  bf16x8 hi = *(const bf16x8*)(p + k0 + 16);
  return __builtin_shufflevector(lo, hi, 0,1,2,3,4,5,6,7,8,9,10,11,12,13,14,15);
}

__device__ __forceinline__ const float* a_base_f32(const float* X, int ld,
                                                   int row0, int lane) {
  return X + (size_t)(row0 + (lane & 15)) * ld + (lane >> 4) * 8;
}
__device__ __forceinline__ bf16x16 a_frag_f32(const float* p, int k0) {
  f32x4 u0 = *(const f32x4*)(p + k0);
  f32x4 u1 = *(const f32x4*)(p + k0 + 4);
  f32x4 u2 = *(const f32x4*)(p + k0 + 16);
  f32x4 u3 = *(const f32x4*)(p + k0 + 20);
  bf16x16 a;
#pragma unroll
  for (int i = 0; i < 4; ++i) {
    a[i]      = (__bf16)u0[i];
    a[4 + i]  = (__bf16)u1[i];
    a[8 + i]  = (__bf16)u2[i];
    a[12 + i] = (__bf16)u3[i];
  }
  return a;
}

__device__ __forceinline__ const float* b_base_f32(const float* X, int ld,
                                                   int n0, int lane) {
  return X + (size_t)(n0 + (lane & 15)) * ld + (lane >> 4) * 8;
}
__device__ __forceinline__ bf16x16 b_frag_f32(const float* p, int k0) {
  f32x4 u0 = *(const f32x4*)(p + k0);
  f32x4 u1 = *(const f32x4*)(p + k0 + 4);
  f32x4 u2 = *(const f32x4*)(p + k0 + 16);
  f32x4 u3 = *(const f32x4*)(p + k0 + 20);
  bf16x16 b;
#pragma unroll
  for (int i = 0; i < 4; ++i) {
    b[i]      = (__bf16)u0[i];
    b[4 + i]  = (__bf16)u1[i];
    b[8 + i]  = (__bf16)u2[i];
    b[12 + i] = (__bf16)u3[i];
  }
  return b;
}

// ---------------------------------------------------------------------------
// Kernel 1: fused Q/K/V projections.  y = x @ W^T + b (bf16 WMMA, f32 acc).
// Q stored [B,H,S,64] pre-scaled by 1/sqrt(64); K stored [B,H,S,64];
// V stored transposed [B,H,64,S] so PV B-operands are contiguous.
// One wave computes a 16(M) x 64(N) strip.  Waves = 3 * 256 * 16 = 12288.
// Software-pipelined: loads for k0+32 issue before the WMMAs consuming k0.
// ---------------------------------------------------------------------------
__global__ __launch_bounds__(256) void k_proj(
    const float* __restrict__ xq, const float* __restrict__ xk, const float* __restrict__ xv,
    const float* __restrict__ wq, const float* __restrict__ wk, const float* __restrict__ wv,
    const float* __restrict__ bq, const float* __restrict__ bk, const float* __restrict__ bv,
    __bf16* __restrict__ qh, __bf16* __restrict__ kh, __bf16* __restrict__ vT)
{
  const int MT = ROWS / 16;          // 256
  const int NT = D_MODEL / 64;       // 16
  int wid  = blockIdx.x * 8 + (threadIdx.x >> 5);
  int lane = threadIdx.x & 31;
  int mat  = wid / (MT * NT);        // 0=Q 1=K 2=V
  int rem  = wid % (MT * NT);
  int m0   = (rem / NT) * 16;
  int n0   = (rem % NT) * 64;

  const float* X  = (mat == 0) ? xq : (mat == 1) ? xk : xv;
  const float* W  = (mat == 0) ? wq : (mat == 1) ? wk : wv;
  const float* Bp = (mat == 0) ? bq : (mat == 1) ? bk : bv;

  const float* pa  = a_base_f32(X, D_MODEL, m0, lane);
  const float* pb0 = b_base_f32(W, D_MODEL, n0,      lane);
  const float* pb1 = b_base_f32(W, D_MODEL, n0 + 16, lane);
  const float* pb2 = b_base_f32(W, D_MODEL, n0 + 32, lane);
  const float* pb3 = b_base_f32(W, D_MODEL, n0 + 48, lane);

  f32x8 acc0 = {}, acc1 = {}, acc2 = {}, acc3 = {};
  bf16x16 a  = a_frag_f32(pa, 0);
  bf16x16 b0 = b_frag_f32(pb0, 0);
  bf16x16 b1 = b_frag_f32(pb1, 0);
  bf16x16 b2 = b_frag_f32(pb2, 0);
  bf16x16 b3 = b_frag_f32(pb3, 0);

#pragma unroll 4
  for (int k0 = 0; k0 < D_MODEL - 32; k0 += 32) {
    // prefetch next k-slice before consuming the current one
    bf16x16 an  = a_frag_f32(pa, k0 + 32);
    bf16x16 bn0 = b_frag_f32(pb0, k0 + 32);
    bf16x16 bn1 = b_frag_f32(pb1, k0 + 32);
    bf16x16 bn2 = b_frag_f32(pb2, k0 + 32);
    bf16x16 bn3 = b_frag_f32(pb3, k0 + 32);
    acc0 = wmma_bf16(a, b0, acc0);
    acc1 = wmma_bf16(a, b1, acc1);
    acc2 = wmma_bf16(a, b2, acc2);
    acc3 = wmma_bf16(a, b3, acc3);
    a = an; b0 = bn0; b1 = bn1; b2 = bn2; b3 = bn3;
  }
  acc0 = wmma_bf16(a, b0, acc0);
  acc1 = wmma_bf16(a, b1, acc1);
  acc2 = wmma_bf16(a, b2, acc2);
  acc3 = wmma_bf16(a, b3, acc3);

  f32x8 acc[4] = {acc0, acc1, acc2, acc3};
  int half = lane >> 4;
  int nlo  = lane & 15;
#pragma unroll
  for (int f = 0; f < 4; ++f) {
    int c    = n0 + f * 16 + nlo;            // output column
    float bi = Bp[c];
    int hh = c >> 6;                         // head
    int dd = c & 63;                         // within-head dim
#pragma unroll
    for (int v = 0; v < 8; ++v) {
      int r  = m0 + v + half * 8;            // output row
      int bb = r >> 11;                      // batch
      int ss = r & 2047;                     // sequence pos
      float y = acc[f][v] + bi;
      if (mat == 0) {
        qh[(((size_t)(bb * N_HEADS + hh)) * SEQ + ss) * D_K + dd] = (__bf16)(y * 0.125f);
      } else if (mat == 1) {
        kh[(((size_t)(bb * N_HEADS + hh)) * SEQ + ss) * D_K + dd] = (__bf16)y;
      } else {
        vT[(((size_t)(bb * N_HEADS + hh)) * D_K + dd) * SEQ + ss] = (__bf16)y;
      }
    }
  }
}

// ---------------------------------------------------------------------------
// Kernel 2: streaming (flash) attention.  One wave owns 16 query rows.
// Scores computed transposed: ST = K_tile @ Q^T, so lane holds column s and the
// exp'd probabilities repack into the PV A-operand with no lane shuffles.
// V-fragment loads and next-chunk K loads issue before the softmax VALU chain
// so memory latency hides behind it.  Waves = B*H*(S/16) = 4096.
// ---------------------------------------------------------------------------
__global__ __launch_bounds__(256) void k_attn(
    const __bf16* __restrict__ qh, const __bf16* __restrict__ kh,
    const __bf16* __restrict__ vT, __bf16* __restrict__ ao)
{
  int wid  = blockIdx.x * 8 + (threadIdx.x >> 5);
  int lane = threadIdx.x & 31;
  int st   = wid & 127;                 // s-tile (128 per head)
  int bh   = wid >> 7;                  // 0..31
  int bb   = bh >> 4, hh = bh & 15;
  int s0   = st * 16;

  const __bf16* Q = qh + (size_t)bh * SEQ * D_K;   // [2048][64]
  const __bf16* K = kh + (size_t)bh * SEQ * D_K;   // [2048][64]
  const __bf16* V = vT + (size_t)bh * D_K * SEQ;   // [64][2048]

  // Q^T as B-operands (fixed for the whole wave): B[d][s] = Q[s][d]
  const __bf16* pq = b_base_bf16(Q, D_K, s0, lane);
  bf16x16 bq0 = b_frag_bf16(pq, 0);
  bf16x16 bq1 = b_frag_bf16(pq, 32);

  // Per-lane base pointers: K A-frags at offset t*64 + d0 (+16 rows = +1024)
  const __bf16* pk  = a_base_bf16(K, D_K, 0, lane);
  const __bf16* pv0 = b_base_bf16(V, SEQ, 0,  lane);
  const __bf16* pv1 = b_base_bf16(V, SEQ, 16, lane);
  const __bf16* pv2 = b_base_bf16(V, SEQ, 32, lane);
  const __bf16* pv3 = b_base_bf16(V, SEQ, 48, lane);

  f32x8 o0 = {}, o1 = {}, o2 = {}, o3 = {};
  float mrun = -3.0e38f, lrun = 0.0f;

  bf16x16 ka0 = a_frag_bf16(pk, 0);
  bf16x16 ka1 = a_frag_bf16(pk, 32);
  bf16x16 kb0 = a_frag_bf16(pk, 1024);
  bf16x16 kb1 = a_frag_bf16(pk, 1024 + 32);

  for (int t0 = 0; t0 < SEQ; t0 += 32) {
    // ST tiles (t rows x s cols), d reduced over two K=32 WMMAs
    f32x8 st1 = {}, st2 = {};
    st1 = wmma_bf16(ka0, bq0, st1);
    st1 = wmma_bf16(ka1, bq1, st1);
    st2 = wmma_bf16(kb0, bq0, st2);
    st2 = wmma_bf16(kb1, bq1, st2);

    // Issue V loads for this chunk and K loads for the next chunk NOW, so they
    // are in flight during the softmax VALU chain below.
    bf16x16 vb0 = b_frag_bf16(pv0, t0);
    bf16x16 vb1 = b_frag_bf16(pv1, t0);
    bf16x16 vb2 = b_frag_bf16(pv2, t0);
    bf16x16 vb3 = b_frag_bf16(pv3, t0);
    int tn = (t0 + 32 < SEQ) ? (t0 + 32) * D_K : 0;   // clamp tail (values unused)
    ka0 = a_frag_bf16(pk, tn);
    ka1 = a_frag_bf16(pk, tn + 32);
    kb0 = a_frag_bf16(pk, tn + 1024);
    kb1 = a_frag_bf16(pk, tn + 1024 + 32);

    // --- online softmax over this 32-wide t chunk (per s-row = lane%16) ---
    float cmax = -3.0e38f;
#pragma unroll
    for (int v = 0; v < 8; ++v) {
      cmax = fmaxf(cmax, st1[v]);
      cmax = fmaxf(cmax, st2[v]);
    }
    cmax = fmaxf(cmax, lane_bcast(cmax, lane ^ 16));   // pair holds same s
    float mnew  = fmaxf(mrun, cmax);
    float alpha = __expf(mrun - mnew);

    float csum = 0.0f;
    bf16x16 pa;                                        // P as A-operand (16x32)
#pragma unroll
    for (int v = 0; v < 8; ++v) {
      float p1 = __expf(st1[v] - mnew);
      float p2 = __expf(st2[v] - mnew);
      csum += p1 + p2;
      pa[v]     = (__bf16)p1;                          // k = (lane/16)*8 + v
      pa[8 + v] = (__bf16)p2;                          // k = 16 + (lane/16)*8 + v
    }
    csum += lane_bcast(csum, lane ^ 16);
    lrun = lrun * alpha + csum;
    mrun = mnew;

    // rescale running output: O-frag VGPR v holds row m = v + 8*(lane/16)
    int base = (lane >> 4) * 8;
#pragma unroll
    for (int v = 0; v < 8; ++v) {
      float av = lane_bcast(alpha, base + v);
      o0[v] *= av; o1[v] *= av; o2[v] *= av; o3[v] *= av;
    }

    // O += P @ V  (B[t][d] = vT[d][t])
    o0 = wmma_bf16(pa, vb0, o0);
    o1 = wmma_bf16(pa, vb1, o1);
    o2 = wmma_bf16(pa, vb2, o2);
    o3 = wmma_bf16(pa, vb3, o3);
  }

  // normalize and store into ao as [B, S, D_MODEL] bf16
  f32x8 o[4] = {o0, o1, o2, o3};
  float inv = 1.0f / lrun;
  int base = (lane >> 4) * 8;
  int nlo  = lane & 15;
#pragma unroll
  for (int v = 0; v < 8; ++v) {
    float iv = lane_bcast(inv, base + v);
    int s = s0 + base + v;
    size_t off = ((size_t)bb * SEQ + s) * D_MODEL + hh * D_K;
#pragma unroll
    for (int f = 0; f < 4; ++f) {
      ao[off + f * 16 + nlo] = (__bf16)(o[f][v] * iv);
    }
  }
}

// ---------------------------------------------------------------------------
// Kernel 3: output projection  out = ao @ w_o^T + b_o  (f32 result).
// Software-pipelined like k_proj.  Waves = 256 * 16 = 4096.
// ---------------------------------------------------------------------------
__global__ __launch_bounds__(256) void k_out(
    const __bf16* __restrict__ ao, const float* __restrict__ wo,
    const float* __restrict__ bo, float* __restrict__ out)
{
  const int NT = D_MODEL / 64;      // 16
  int wid  = blockIdx.x * 8 + (threadIdx.x >> 5);
  int lane = threadIdx.x & 31;
  int m0   = (wid / NT) * 16;
  int n0   = (wid % NT) * 64;

  const __bf16* pa  = a_base_bf16(ao, D_MODEL, m0, lane);
  const float*  pb0 = b_base_f32(wo, D_MODEL, n0,      lane);
  const float*  pb1 = b_base_f32(wo, D_MODEL, n0 + 16, lane);
  const float*  pb2 = b_base_f32(wo, D_MODEL, n0 + 32, lane);
  const float*  pb3 = b_base_f32(wo, D_MODEL, n0 + 48, lane);

  f32x8 acc0 = {}, acc1 = {}, acc2 = {}, acc3 = {};
  bf16x16 a  = a_frag_bf16(pa, 0);
  bf16x16 b0 = b_frag_f32(pb0, 0);
  bf16x16 b1 = b_frag_f32(pb1, 0);
  bf16x16 b2 = b_frag_f32(pb2, 0);
  bf16x16 b3 = b_frag_f32(pb3, 0);

#pragma unroll 4
  for (int k0 = 0; k0 < D_MODEL - 32; k0 += 32) {
    bf16x16 an  = a_frag_bf16(pa, k0 + 32);
    bf16x16 bn0 = b_frag_f32(pb0, k0 + 32);
    bf16x16 bn1 = b_frag_f32(pb1, k0 + 32);
    bf16x16 bn2 = b_frag_f32(pb2, k0 + 32);
    bf16x16 bn3 = b_frag_f32(pb3, k0 + 32);
    acc0 = wmma_bf16(a, b0, acc0);
    acc1 = wmma_bf16(a, b1, acc1);
    acc2 = wmma_bf16(a, b2, acc2);
    acc3 = wmma_bf16(a, b3, acc3);
    a = an; b0 = bn0; b1 = bn1; b2 = bn2; b3 = bn3;
  }
  acc0 = wmma_bf16(a, b0, acc0);
  acc1 = wmma_bf16(a, b1, acc1);
  acc2 = wmma_bf16(a, b2, acc2);
  acc3 = wmma_bf16(a, b3, acc3);

  f32x8 acc[4] = {acc0, acc1, acc2, acc3};
  int half = lane >> 4;
  int nlo  = lane & 15;
#pragma unroll
  for (int f = 0; f < 4; ++f) {
    int c    = n0 + f * 16 + nlo;
    float bi = bo[c];
#pragma unroll
    for (int v = 0; v < 8; ++v) {
      int r = m0 + v + half * 8;
      out[(size_t)r * D_MODEL + c] = acc[f][v] + bi;
    }
  }
}

// ---------------------------------------------------------------------------
extern "C" void kernel_launch(void* const* d_in, const int* in_sizes, int n_in,
                              void* d_out, int out_size, void* d_ws, size_t ws_size,
                              hipStream_t stream) {
  (void)in_sizes; (void)n_in; (void)out_size; (void)ws_size;
  const float* q  = (const float*)d_in[0];
  const float* k  = (const float*)d_in[1];
  const float* v  = (const float*)d_in[2];
  const float* wq = (const float*)d_in[3];
  const float* bq = (const float*)d_in[4];
  const float* wk = (const float*)d_in[5];
  const float* bk = (const float*)d_in[6];
  const float* wv = (const float*)d_in[7];
  const float* bv = (const float*)d_in[8];
  const float* wo = (const float*)d_in[9];
  const float* bo = (const float*)d_in[10];
  float* out = (float*)d_out;

  // workspace: 4 bf16 matrices of 4096x1024 (8 MB each, 32 MB total)
  const size_t N = (size_t)ROWS * D_MODEL;
  __bf16* ws = (__bf16*)d_ws;
  __bf16* qh = ws;
  __bf16* kh = ws + N;
  __bf16* vT = ws + 2 * N;
  __bf16* ao = ws + 3 * N;

  // 12288 waves / 8 per block
  k_proj<<<1536, 256, 0, stream>>>(q, k, v, wq, wk, wv, bq, bk, bv, qh, kh, vT);
  // 4096 waves / 8 per block
  k_attn<<<512, 256, 0, stream>>>(qh, kh, vT, ao);
  // 4096 waves / 8 per block
  k_out<<<512, 256, 0, stream>>>(ao, wo, bo, out);
}
